// NoisyTopkRouter_71528385347886
// MI455X (gfx1250) — matrix operand order repaired and checked by reading, not compiled, not run on hardware
//
#include <hip/hip_runtime.h>
#include <math.h>

typedef __attribute__((ext_vector_type(16))) _Float16 v16h;
typedef __attribute__((ext_vector_type(8)))  float    v8f;

#define D_DIM 1024
#define E_DIM 64
#define KCHUNK 128
#define KSTEP 32
#define WROW (KCHUNK + 16)      // 144 halves = 288B row pitch: every fragment 32B-aligned -> ds_load_b128
#define WMAT (E_DIM * WROW)     // halves per matrix slice (9216)
#define THREADS 128
#define WAVES 4
#define TOK_PER_BLOCK 128       // 4 waves x 2 tiles x 16 tokens
#define NOISY_ROWS 64           // epilogue processed in two 64-token passes
#define NOISY_STRIDE (E_DIM + 1)

union SmemU {
    _Float16 w[2 * WMAT];                      // 36,864 B (GEMM phase)
    float    noisy[NOISY_ROWS * NOISY_STRIDE]; // 16,640 B (epilogue phase)
};

__global__ __launch_bounds__(THREADS) void noisy_topk_router_kernel(
    const float* __restrict__ h, const float* __restrict__ Ww,
    const float* __restrict__ bw, const float* __restrict__ Wn,
    const float* __restrict__ bn, const float* __restrict__ noise,
    float* __restrict__ out_sparse, int* __restrict__ out_ix,
    float* __restrict__ out_full)
{
    __shared__ SmemU sm;

    const int tid  = threadIdx.x;
    const int wave = tid >> 5;           // wave32
    const int lane = tid & 31;
    const int lhi  = lane >> 4;          // half-wave select
    const int llo  = lane & 15;
    const int tile_base = blockIdx.x * TOK_PER_BLOCK + wave * 32;  // 2 tiles (32 tokens) per wave

    v8f accw[2][4], accn[2][4];
#pragma unroll
    for (int t = 0; t < 2; ++t)
#pragma unroll
        for (int nt = 0; nt < 4; ++nt) {
            accw[t][nt] = (v8f){0.f,0.f,0.f,0.f,0.f,0.f,0.f,0.f};
            accn[t][nt] = (v8f){0.f,0.f,0.f,0.f,0.f,0.f,0.f,0.f};
        }

    // A rows (contiguous k order, same as B): lane -> token tile_base + llo (+16 for tile 1), K base lhi*16
    const float* hbase0 = h + (size_t)(tile_base + llo) * D_DIM + lhi * 16;
    const float* hbase1 = hbase0 + (size_t)16 * D_DIM;
    // B fragments: lane -> expert column llo (+16*nt), K halves lhi*16; single base + const offsets
    const _Float16* bbase = sm.w + llo * WROW + lhi * 16;

    for (int kc = 0; kc < D_DIM / KCHUNK; ++kc) {
        __syncthreads();
        // ---- cooperative load of W_w/W_n K-chunk, f32->f16, transposed [expert][k] ----
        {
            const int k0 = kc * KCHUNK;
#pragma unroll
            for (int j = 0; j < 16; ++j) {
                const int idx = (j * THREADS + tid) * 4;  // 0..8188 float4 granule over 128x64 slice
                const int kl  = idx >> 6;
                const int n   = idx & 63;
                const float4 ww = *(const float4*)(Ww + (size_t)(k0 + kl) * E_DIM + n);
                const float4 wn = *(const float4*)(Wn + (size_t)(k0 + kl) * E_DIM + n);
                sm.w[(n + 0) * WROW + kl] = (_Float16)ww.x;
                sm.w[(n + 1) * WROW + kl] = (_Float16)ww.y;
                sm.w[(n + 2) * WROW + kl] = (_Float16)ww.z;
                sm.w[(n + 3) * WROW + kl] = (_Float16)ww.w;
                sm.w[WMAT + (n + 0) * WROW + kl] = (_Float16)wn.x;
                sm.w[WMAT + (n + 1) * WROW + kl] = (_Float16)wn.y;
                sm.w[WMAT + (n + 2) * WROW + kl] = (_Float16)wn.z;
                sm.w[WMAT + (n + 3) * WROW + kl] = (_Float16)wn.w;
            }
        }
        __syncthreads();

        // ---- WMMA over this K-chunk: each B fragment feeds 2 token tiles ----
#pragma unroll
        for (int kk = 0; kk < KCHUNK / KSTEP; ++kk) {
            const float* hp0 = hbase0 + kc * KCHUNK + kk * KSTEP;
            const float* hp1 = hbase1 + kc * KCHUNK + kk * KSTEP;
            const float4 a00 = *(const float4*)(hp0 +  0);
            const float4 a01 = *(const float4*)(hp0 +  4);
            const float4 a02 = *(const float4*)(hp0 +  8);
            const float4 a03 = *(const float4*)(hp0 + 12);
            const float4 a10 = *(const float4*)(hp1 +  0);
            const float4 a11 = *(const float4*)(hp1 +  4);
            const float4 a12 = *(const float4*)(hp1 +  8);
            const float4 a13 = *(const float4*)(hp1 + 12);
            v16h a0, a1;
            a0[0]=(_Float16)a00.x; a0[1]=(_Float16)a00.y; a0[2]=(_Float16)a00.z; a0[3]=(_Float16)a00.w;
            a0[4]=(_Float16)a01.x; a0[5]=(_Float16)a01.y; a0[6]=(_Float16)a01.z; a0[7]=(_Float16)a01.w;
            a0[8]=(_Float16)a02.x; a0[9]=(_Float16)a02.y; a0[10]=(_Float16)a02.z; a0[11]=(_Float16)a02.w;
            a0[12]=(_Float16)a03.x; a0[13]=(_Float16)a03.y; a0[14]=(_Float16)a03.z; a0[15]=(_Float16)a03.w;
            a1[0]=(_Float16)a10.x; a1[1]=(_Float16)a10.y; a1[2]=(_Float16)a10.z; a1[3]=(_Float16)a10.w;
            a1[4]=(_Float16)a11.x; a1[5]=(_Float16)a11.y; a1[6]=(_Float16)a11.z; a1[7]=(_Float16)a11.w;
            a1[8]=(_Float16)a12.x; a1[9]=(_Float16)a12.y; a1[10]=(_Float16)a12.z; a1[11]=(_Float16)a12.w;
            a1[12]=(_Float16)a13.x; a1[13]=(_Float16)a13.y; a1[14]=(_Float16)a13.z; a1[15]=(_Float16)a13.w;

#pragma unroll
            for (int nt = 0; nt < 4; ++nt) {
                const v16h bwf = *(const v16h*)(bbase + nt * 16 * WROW + kk * KSTEP);
                const v16h bnf = *(const v16h*)(bbase + WMAT + nt * 16 * WROW + kk * KSTEP);
                accw[0][nt] = __builtin_amdgcn_wmma_f32_16x16x32_f16(
                    false, a0, false, bwf, (short)0, accw[0][nt], false, false);
                accw[1][nt] = __builtin_amdgcn_wmma_f32_16x16x32_f16(
                    false, a1, false, bwf, (short)0, accw[1][nt], false, false);
                accn[0][nt] = __builtin_amdgcn_wmma_f32_16x16x32_f16(
                    false, a0, false, bnf, (short)0, accn[0][nt], false, false);
                accn[1][nt] = __builtin_amdgcn_wmma_f32_16x16x32_f16(
                    false, a1, false, bnf, (short)0, accn[1][nt], false, false);
            }
        }
    }

    // ---- fused epilogue: two 64-token passes (noisy staging + top-2 / softmaxes) ----
#pragma unroll
    for (int half = 0; half < 2; ++half) {
        __syncthreads();   // W buffer (or previous noisy pass) dead
        const int tbase_h = tile_base + half * 16;
#pragma unroll
        for (int nt = 0; nt < 4; ++nt) {
            const int e = nt * 16 + llo;
            const float bwe = bw[e];
            const float bne = bn[e];
#pragma unroll
            for (int r = 0; r < 8; ++r) {
                const int mloc = r + 8 * lhi;           // C/D layout: VGPR r, half-wave selects M
                const int trow = wave * 16 + mloc;      // 0..63
                const float nz = noise[(size_t)(tbase_h + mloc) * E_DIM + e];
                const float lw = accw[half][nt][r] + bwe;
                const float ln = accn[half][nt][r] + bne;
                const float sp = (ln > 20.f) ? ln : log1pf(__expf(ln));
                sm.noisy[trow * NOISY_STRIDE + e] = lw + nz * sp;
            }
        }
        __syncthreads();

        if (tid < NOISY_ROWS) {
            const int token = blockIdx.x * TOK_PER_BLOCK + (tid >> 4) * 32 + half * 16 + (tid & 15);
            float* row = &sm.noisy[tid * NOISY_STRIDE];

            float v1 = -INFINITY, v2 = -INFINITY;
            int   i1 = 0,          i2 = 0;
            for (int e = 0; e < E_DIM; ++e) {
                const float x = row[e];
                if (x > v1)      { v2 = v1; i2 = i1; v1 = x; i1 = e; }
                else if (x > v2) { v2 = x;  i2 = e; }
            }

            float sum = 0.f;
            for (int e = 0; e < E_DIM; ++e) {
                const float ex = __expf(row[e] - v1);
                sum += ex;
                row[e] = ex;
            }
            const float inv = 1.f / sum;

            float* fullp = out_full + (size_t)token * E_DIM;
            for (int e = 0; e < E_DIM; e += 4) {
                float4 o = { row[e] * inv, row[e+1] * inv, row[e+2] * inv, row[e+3] * inv };
                *(float4*)(fullp + e) = o;
            }

            const float p1 = 1.f / (1.f + __expf(v2 - v1));
            const float p2 = 1.f - p1;
            float* sp = out_sparse + (size_t)token * E_DIM;
            const float4 z = {0.f, 0.f, 0.f, 0.f};
            for (int e = 0; e < E_DIM; e += 4) *(float4*)(sp + e) = z;
            sp[i1] = p1;
            sp[i2] = p2;

            out_ix[token * 2 + 0] = i1;
            out_ix[token * 2 + 1] = i2;
        }
    }
}

extern "C" void kernel_launch(void* const* d_in, const int* in_sizes, int n_in,
                              void* d_out, int out_size, void* d_ws, size_t ws_size,
                              hipStream_t stream) {
    const float* h     = (const float*)d_in[0];
    const float* Ww    = (const float*)d_in[1];
    const float* bwv   = (const float*)d_in[2];
    const float* Wn    = (const float*)d_in[3];
    const float* bnv   = (const float*)d_in[4];
    const float* noise = (const float*)d_in[5];

    const int T = in_sizes[5] / E_DIM;   // B*S tokens (32768)

    float* out        = (float*)d_out;
    float* out_sparse = out;                                   // [T,64] f32
    int*   out_ix     = (int*)(out + (size_t)T * E_DIM);       // [T,2]  i32 (same width as f32)
    float* out_full   = out + (size_t)T * E_DIM + (size_t)T*2; // [T,64] f32

    const int grid = (T + TOK_PER_BLOCK - 1) / TOK_PER_BLOCK;  // 256 blocks
    noisy_topk_router_kernel<<<grid, THREADS, 0, stream>>>(
        h, Ww, bwv, Wn, bnv, noise, out_sparse, out_ix, out_full);
}
